// VelocityEGNNLayer_13950053777588
// MI455X (gfx1250) — matrix operand order repaired and checked by reading, not compiled, not run on hardware
//
#include <hip/hip_runtime.h>

#define H 128
#define NN 512
#define TI 16
#define TJ 16
#define THREADS 256

typedef _Float16 half8  __attribute__((ext_vector_type(8)));
typedef _Float16 half16 __attribute__((ext_vector_type(16)));
typedef float    v8f    __attribute__((ext_vector_type(8)));
typedef unsigned int uint32x4 __attribute__((ext_vector_type(4)));
typedef int      int32x4 __attribute__((ext_vector_type(4)));
typedef int      int32x8 __attribute__((ext_vector_type(8)));

__device__ __forceinline__ float fast_rcp(float x) { return __builtin_amdgcn_rcpf(x); }
__device__ __forceinline__ float silu_f(float x) { return x * fast_rcp(1.0f + __expf(-x)); }
__device__ __forceinline__ float sigmoid_f(float x) { return fast_rcp(1.0f + __expf(-x)); }

__device__ __forceinline__ half16 cat8(half8 lo, half8 hi) {
  return __builtin_shufflevector(lo, hi, 0,1,2,3,4,5,6,7,8,9,10,11,12,13,14,15);
}

// A fragment (16x32 f16, M=edges, K): lane&15 = row, lane>>4 selects K-halves.
__device__ __forceinline__ half16 load_a_frag(const _Float16* base, int lane, int kc) {
  const _Float16* p = base + (lane & 15) * H + kc * 32 + (lane >> 4) * 8;
  half8 lo = *(const half8*)(p);
  half8 hi = *(const half8*)(p + 16);
  return cat8(lo, hi);
}

// B fragment (32x16 f16, KxN) from transposed weights sW[out][in]:
__device__ __forceinline__ half16 load_b_frag(const _Float16* sW, int lane, int kc, int oT) {
  const _Float16* p = sW + (oT * 16 + (lane & 15)) * H + kc * 32 + (lane >> 4) * 8;
  half8 lo = *(const half8*)(p);
  half8 hi = *(const half8*)(p + 16);
  return cat8(lo, hi);
}

// Tensor Data Mover: async DMA of one 128x128 f16 weight matrix (1-D, 16384
// elements, data_size=2B) from global to LDS. Issue from one wave only;
// completion fenced with s_wait_tensorcnt + workgroup barrier.
__device__ __forceinline__ void tdm_load_w(const _Float16* gsrc, _Float16* sdst) {
  unsigned long long ga = (unsigned long long)(uintptr_t)gsrc;
  unsigned int lds = (unsigned int)(uintptr_t)sdst;   // low 32 bits = LDS offset
  uint32x4 g0;
  g0[0] = 1u;                                          // count=1 (valid user D#)
  g0[1] = lds;                                         // lds_addr
  g0[2] = (unsigned int)(ga & 0xFFFFFFFFu);            // global_addr[31:0]
  g0[3] = (unsigned int)((ga >> 32) & 0x01FFFFFFu) | (2u << 30);  // [56:32], type=2
  int32x8 g1;
  g1[0] = (int)(1u << 16);        // workgroup_mask=0, data_size=1 (2 bytes)
  g1[1] = (int)(16384u << 16);    // tensor_dim0[15:0] in bits [63:48]
  g1[2] = (int)(1u << 16);        // tensor_dim0 hi=0, tensor_dim1=1
  g1[3] = (int)(16384u << 16);    // tile_dim0 = 16384 in bits [127:112]
  g1[4] = 0;                      // tile_dim1=0 (unused), tile_dim2=0
  g1[5] = 16384;                  // tensor_dim0_stride
  g1[6] = 0;
  g1[7] = 0;
  int32x4 z4 = {0, 0, 0, 0};
  int32x8 z8 = {0, 0, 0, 0, 0, 0, 0, 0};
  __builtin_amdgcn_tensor_load_to_lds(g0, g1, z4, z4, z8, 0);
}

// GEMM over edges (M) x out (N), K=128, epilogue: silu(.+bias)*wout summed per edge.
__device__ __forceinline__ void gemm_dot_phase(const _Float16* sM, const _Float16* sW,
                                               const float* biasv, const float* wout,
                                               float* sdot, int lane, int w) {
  for (int oT = 0; oT < 8; ++oT) {
    v8f c0 = {}, c1 = {};
#pragma unroll
    for (int kc = 0; kc < 4; ++kc) {
      half16 bf = load_b_frag(sW, lane, kc, oT);
      half16 a0 = load_a_frag(sM + (2 * w) * 16 * H, lane, kc);
      half16 a1 = load_a_frag(sM + (2 * w + 1) * 16 * H, lane, kc);
      c0 = __builtin_amdgcn_wmma_f32_16x16x32_f16(false, a0, false, bf, (short)0, c0, false, false);
      c1 = __builtin_amdgcn_wmma_f32_16x16x32_f16(false, a1, false, bf, (short)0, c1, false, false);
    }
    int o = oT * 16 + (lane & 15);
    float bb = biasv[o], w2 = wout[o];
#pragma unroll
    for (int r = 0; r < 8; ++r) {
      int m = r + ((lane >> 4) << 3);
      atomicAdd(&sdot[(2 * w) * 16 + m], silu_f(c0[r] + bb) * w2);
      atomicAdd(&sdot[(2 * w + 1) * 16 + m], silu_f(c1[r] + bb) * w2);
    }
  }
}

// ---------------------------------------------------------------------------
// Prep: Ahat = h @ We1[0:128], Bhat = h @ We1[128:256]
__global__ __launch_bounds__(128)
void egnn_prep_kernel(const float* __restrict__ h, const float* __restrict__ We1,
                      float* __restrict__ Ahat, float* __restrict__ Bhat) {
  __shared__ float sh[H];
  int n = blockIdx.x, t = threadIdx.x;
  sh[t] = h[n * H + t];
  __syncthreads();
  float fa = 0.f, fb = 0.f;
  for (int k = 0; k < H; ++k) {
    float hv = sh[k];
    fa += hv * We1[k * H + t];
    fb += hv * We1[(H + k) * H + t];
  }
  Ahat[n * H + t] = fa;
  Bhat[n * H + t] = fb;
}

// Transpose+convert weights to f16 [out][in]
__global__ __launch_bounds__(128)
void egnn_wt_kernel(const float* __restrict__ We2, const float* __restrict__ Wp1,
                    const float* __restrict__ Wv1,
                    _Float16* __restrict__ WeT2, _Float16* __restrict__ WpT,
                    _Float16* __restrict__ WvT) {
  int o = blockIdx.x, k = threadIdx.x;
  WeT2[o * H + k] = (_Float16)We2[k * H + o];
  WpT[o * H + k]  = (_Float16)Wp1[k * H + o];
  WvT[o * H + k]  = (_Float16)Wv1[k * H + o];
}

// ---------------------------------------------------------------------------
// Main edge kernel: one block per (b, i-tile16, j-tile16)
__global__ __launch_bounds__(THREADS)
void egnn_edge_kernel(const float* __restrict__ pos, const float* __restrict__ vel,
                      const float* __restrict__ We1, const float* __restrict__ be1,
                      const float* __restrict__ be2, const float* __restrict__ Wa,
                      const float* __restrict__ ba,
                      const float* __restrict__ bp1, const float* __restrict__ Wp2,
                      const float* __restrict__ bv1, const float* __restrict__ Wv2,
                      const float* __restrict__ Ahat, const float* __restrict__ Bhat,
                      const _Float16* __restrict__ WeT2, const _Float16* __restrict__ WpT,
                      const _Float16* __restrict__ WvT,
                      float* __restrict__ maggAcc, float* __restrict__ posAcc,
                      float* __restrict__ velAcc) {
  __shared__ __align__(16) _Float16 sW0[H * H];       // We2, later Wv1 (TDM-filled)
  __shared__ __align__(16) _Float16 sW1[H * H];       // Wp1 (TDM-filled)
  __shared__ __align__(16) _Float16 sM[TI * TJ * H];  // 64 KB edge activations
  __shared__ float sAi[TI * H], sBj[TJ * H];
  __shared__ float sPi[TI * 3], sPj[TJ * 3], sVi[TI * 3], sVj[TJ * 3];
  __shared__ float sBiasA[H], sBiasB[H], sBiasC[H];
  __shared__ float sdotA[TI * TJ], sdotP[TI * TJ];
  __shared__ float smagg[TI * H];
  __shared__ float sposacc[TI * 3], svelacc[TI * 3];

  const int tid = threadIdx.x;
  const int lane = tid & 31;
  const int w = tid >> 5;
  const int jt = blockIdx.x, it = blockIdx.y, b = blockIdx.z;
  const int i0 = it * TI, j0 = jt * TJ;
  const int nodeI = b * NN + i0;
  const int nodeJ = b * NN + j0;

  // ---- kick off async weight DMAs (wave 0 only; uniform branch) ----
  if (w == 0) {
    tdm_load_w(WeT2, sW0);   // needed for GEMM2
    tdm_load_w(WpT, sW1);    // needed for pos head; overlaps GEMM2
  }

  // ---- init + staging ----
  if (tid < TI * TJ) { sdotA[tid] = 0.f; sdotP[tid] = 0.f; }
  for (int r = tid; r < TI * H; r += THREADS) smagg[r] = 0.f;
  if (tid < TI * 3) { sposacc[tid] = 0.f; svelacc[tid] = 0.f; }
  for (int r = tid; r < TI * H; r += THREADS) {
    int row = r >> 7, col = r & (H - 1);
    sAi[r] = Ahat[(nodeI + row) * H + col];
    sBj[r] = Bhat[(nodeJ + row) * H + col];
  }
  if (tid < TI * 3) {
    int ii = tid / 3, d = tid - ii * 3;
    sPi[tid] = pos[(nodeI + ii) * 3 + d];
    sVi[tid] = vel[(nodeI + ii) * 3 + d];
    sPj[tid] = pos[(nodeJ + ii) * 3 + d];
    sVj[tid] = vel[(nodeJ + ii) * 3 + d];
  }
  if (tid < H) {
    sBiasA[tid] = be1[tid];
    sBiasB[tid] = We1[256 * H + tid];   // pos_dist_sq weight row
    sBiasC[tid] = We1[257 * H + tid];   // vel_dist_sq weight row
  }
  __syncthreads();

  // ---- stage 1: m = silu(Ahat[i] + Bhat[j] + dp*wp + dv*wv + be1) -> sM f16 ----
  {
    int e = tid, ii = e >> 4, jj = e & 15;
    float dp = 0.f, dv = 0.f;
#pragma unroll
    for (int d = 0; d < 3; ++d) {
      float t1 = sPi[ii * 3 + d] - sPj[jj * 3 + d]; dp += t1 * t1;
      float t2 = sVi[ii * 3 + d] - sVj[jj * 3 + d]; dv += t2 * t2;
    }
    for (int k = 0; k < H; ++k) {
      float x = sAi[ii * H + k] + sBj[jj * H + k] + dp * sBiasB[k] + dv * sBiasC[k] + sBiasA[k];
      sM[e * H + k] = (_Float16)silu_f(x);
    }
  }
  __syncthreads();
  if (tid < H) { sBiasA[tid] = be2[tid]; sBiasB[tid] = Wa[tid]; }
  __builtin_amdgcn_s_wait_tensorcnt((short)1);   // We2 DMA landed (in-order)
  __syncthreads();

  // ---- GEMM2: m_ij = silu(m @ We2 + be2); att dot accumulated into sdotA ----
  v8f acc[16];
#pragma unroll
  for (int oT = 0; oT < 8; ++oT) {
    v8f c0 = {}, c1 = {};
#pragma unroll
    for (int kc = 0; kc < 4; ++kc) {
      half16 bf = load_b_frag(sW0, lane, kc, oT);
      half16 a0 = load_a_frag(sM + (2 * w) * 16 * H, lane, kc);
      half16 a1 = load_a_frag(sM + (2 * w + 1) * 16 * H, lane, kc);
      c0 = __builtin_amdgcn_wmma_f32_16x16x32_f16(false, a0, false, bf, (short)0, c0, false, false);
      c1 = __builtin_amdgcn_wmma_f32_16x16x32_f16(false, a1, false, bf, (short)0, c1, false, false);
    }
    int o = oT * 16 + (lane & 15);
    float b2 = sBiasA[o], wa = sBiasB[o];
#pragma unroll
    for (int r = 0; r < 8; ++r) {
      int m = r + ((lane >> 4) << 3);
      float s0 = silu_f(c0[r] + b2);
      float s1 = silu_f(c1[r] + b2);
      c0[r] = s0; c1[r] = s1;
      atomicAdd(&sdotA[(2 * w) * 16 + m], s0 * wa);
      atomicAdd(&sdotA[(2 * w + 1) * 16 + m], s1 * wa);
    }
    acc[oT * 2]     = c0;
    acc[oT * 2 + 1] = c1;
  }
  __syncthreads();

  // ---- attention gate; also start Wv1 DMA (sW0 reads are done) ----
  if (w == 0) tdm_load_w(WvT, sW0);
  {
    float bav = ba[0];
    sdotA[tid] = sigmoid_f(sdotA[tid] + bav);
  }
  __syncthreads();

  // ---- write back att-scaled m_ij to sM; accumulate m_agg partials ----
#pragma unroll
  for (int t = 0; t < 16; ++t) {
    int oT = t >> 1, p = t & 1;
    int eT = 2 * w + p;
    int o = oT * 16 + (lane & 15);
    v8f c = acc[t];
    float msum = 0.f;
#pragma unroll
    for (int r = 0; r < 8; ++r) {
      int edge = eT * 16 + r + ((lane >> 4) << 3);
      float val = c[r] * sdotA[edge];
      sM[edge * H + o] = (_Float16)val;
      msum += val;
    }
    atomicAdd(&smagg[eT * H + o], msum);   // eT == local i index
  }
  __syncthreads();

  // ---- pos head: p = silu(m_ij @ Wp1 + bp1) . Wp2 -> sdotP ----
  if (tid < H) { sBiasA[tid] = bp1[tid]; sBiasB[tid] = Wp2[tid]; }
  __builtin_amdgcn_s_wait_tensorcnt((short)1);   // Wp1 landed (Wv1 may be in flight)
  __syncthreads();
  gemm_dot_phase(sM, sW1, sBiasA, sBiasB, sdotP, lane, w);
  __syncthreads();
  {
    int e = tid, ii = e >> 4, jj = e & 15;
    float pw = tanhf(sdotP[e]);
    sdotP[e] = 0.f;  // re-zero for vel phase (only this thread touched slot e)
#pragma unroll
    for (int d = 0; d < 3; ++d)
      atomicAdd(&sposacc[ii * 3 + d], (sPi[ii * 3 + d] - sPj[jj * 3 + d]) * pw);
  }
  __syncthreads();

  // ---- vel head ----
  if (tid < H) { sBiasA[tid] = bv1[tid]; sBiasB[tid] = Wv2[tid]; }
  __builtin_amdgcn_s_wait_tensorcnt((short)0);   // Wv1 landed
  __syncthreads();
  gemm_dot_phase(sM, sW0, sBiasA, sBiasB, sdotP, lane, w);
  __syncthreads();
  {
    int e = tid, ii = e >> 4, jj = e & 15;
    float vw = tanhf(sdotP[e]);
#pragma unroll
    for (int d = 0; d < 3; ++d)
      atomicAdd(&svelacc[ii * 3 + d], (sVi[ii * 3 + d] - sVj[jj * 3 + d]) * vw);
  }
  __syncthreads();

  // ---- flush block partials to global accumulators ----
#pragma unroll
  for (int r = 0; r < 8; ++r) {
    int idx = tid * 8 + r;
    int ii = idx >> 7, o = idx & (H - 1);
    atomicAdd(&maggAcc[(nodeI + ii) * H + o], smagg[idx]);
  }
  if (tid < TI * 3) {
    atomicAdd(&posAcc[nodeI * 3 + tid], sposacc[tid]);
    atomicAdd(&velAcc[nodeI * 3 + tid], svelacc[tid]);
  }
}

// ---------------------------------------------------------------------------
// Finalize: node MLP + residual, pos/vel updates with clamp
__global__ __launch_bounds__(128)
void egnn_final_kernel(const float* __restrict__ h, const float* __restrict__ pos,
                       const float* __restrict__ vel,
                       const float* __restrict__ magg, const float* __restrict__ posAcc,
                       const float* __restrict__ velAcc,
                       const float* __restrict__ Wn1, const float* __restrict__ bn1,
                       const float* __restrict__ Wn2, const float* __restrict__ bn2,
                       float* __restrict__ out) {
  __shared__ float sh[H], sm[H], sy[H];
  int n = blockIdx.x, t = threadIdx.x;
  sh[t] = h[n * H + t];
  sm[t] = magg[n * H + t];
  __syncthreads();
  float a1 = bn1[t];
  for (int c = 0; c < H; ++c) a1 += sh[c] * Wn1[c * H + t];
  for (int c = 0; c < H; ++c) a1 += sm[c] * Wn1[(H + c) * H + t];
  sy[t] = silu_f(a1);
  __syncthreads();
  float a2 = bn2[t];
  for (int k = 0; k < H; ++k) a2 += sy[k] * Wn2[k * H + t];
  out[n * H + t] = sh[t] + a2;

  const float inv = 1.0f / 511.0f;
  if (t < 3) {
    float po = pos[n * 3 + t] + posAcc[n * 3 + t] * inv;
    po = fminf(fmaxf(po, -100.0f), 100.0f);
    out[2 * NN * H + n * 3 + t] = po;
    float vo = vel[n * 3 + t] + velAcc[n * 3 + t] * inv;
    vo = fminf(fmaxf(vo, -100.0f), 100.0f);
    out[2 * NN * H + 2 * NN * 3 + n * 3 + t] = vo;
  }
}

// ---------------------------------------------------------------------------
extern "C" void kernel_launch(void* const* d_in, const int* in_sizes, int n_in,
                              void* d_out, int out_size, void* d_ws, size_t ws_size,
                              hipStream_t stream) {
  const float* h   = (const float*)d_in[0];
  const float* pos = (const float*)d_in[1];
  const float* vel = (const float*)d_in[2];
  const float* We1 = (const float*)d_in[3];
  const float* be1 = (const float*)d_in[4];
  const float* We2 = (const float*)d_in[5];
  const float* be2 = (const float*)d_in[6];
  const float* Wa  = (const float*)d_in[7];
  const float* ba  = (const float*)d_in[8];
  const float* Wp1 = (const float*)d_in[9];
  const float* bp1 = (const float*)d_in[10];
  const float* Wp2 = (const float*)d_in[11];
  const float* Wv1 = (const float*)d_in[12];
  const float* bv1 = (const float*)d_in[13];
  const float* Wv2 = (const float*)d_in[14];
  const float* Wn1 = (const float*)d_in[15];
  const float* bn1 = (const float*)d_in[16];
  const float* Wn2 = (const float*)d_in[17];
  const float* bn2 = (const float*)d_in[18];

  float* ws = (float*)d_ws;
  float* magg = ws;                       // 2*512*128  = 131072
  float* posA = ws + 131072;              // 3072
  float* velA = ws + 134144;              // 3072
  float* Ahat = ws + 137216;              // 131072
  float* Bhat = ws + 268288;              // 131072
  _Float16* WeT2 = (_Float16*)(ws + 399360);  // 16384 halves each
  _Float16* WpT  = (_Float16*)(ws + 407552);
  _Float16* WvT  = (_Float16*)(ws + 415744);

  (void)hipMemsetAsync(magg, 0, (size_t)(131072 + 3072 + 3072) * sizeof(float), stream);

  egnn_prep_kernel<<<1024, 128, 0, stream>>>(h, We1, Ahat, Bhat);
  egnn_wt_kernel<<<128, 128, 0, stream>>>(We2, Wp1, Wv1, WeT2, WpT, WvT);

  egnn_edge_kernel<<<dim3(32, 32, 2), THREADS, 0, stream>>>(
      pos, vel, We1, be1, be2, Wa, ba, bp1, Wp2, bv1, Wv2,
      Ahat, Bhat, WeT2, WpT, WvT, magg, posA, velA);

  egnn_final_kernel<<<1024, 128, 0, stream>>>(h, pos, vel, magg, posA, velA,
                                              Wn1, bn1, Wn2, bn2, (float*)d_out);
  (void)in_sizes; (void)n_in; (void)out_size; (void)ws_size;
}
